// PSGNet_40673340293839
// MI455X (gfx1250) — matrix operand order repaired and checked by reading, not compiled, not run on hardware
//
#include <hip/hip_runtime.h>
#include <hip/hip_bf16.h>

typedef __attribute__((ext_vector_type(16))) _Float16 v16h;
typedef __attribute__((ext_vector_type(8)))  _Float16 v8h;
typedef __attribute__((ext_vector_type(8)))  float    v8f;

// ---------------- problem constants ----------------
#define BB   4
#define HH   256
#define NPIX (BB*HH*HH)      // 262144
#define FF   64
#define HID  100
#define M1   4096
#define M2   512
#define E1   65536
#define E2   8192
#define DIN  68              // F + 4
#define DINP 96              // DIN padded to multiple of 32
#define HIDP 128             // HID padded to multiple of 32
#define JSFW 20

// ---------------- workspace layout (floats) ----------------
// accumulator (memset-0 each call) region first:
#define O_X1SUM   0
#define O_C1SUM   (O_X1SUM  + M1*FF)
#define O_MO1SUM  (O_C1SUM  + M1*2)
#define O_CNT1    (O_MO1SUM + M1*2)
#define O_C2SUM   (O_CNT1   + M1)
#define O_MO2SUM  (O_C2SUM  + M2*2)
#define O_CNTP2   (O_MO2SUM + M2*2)
#define O_MSUM1   (O_CNTP2  + M2)
#define O_MCNT1   (O_MSUM1  + M1*FF)
#define O_X2SUM   (O_MCNT1  + M1)
#define O_CNTR2   (O_X2SUM  + M2*FF)
#define O_MSUM2   (O_CNTR2  + M2)
#define O_MCNT2   (O_MSUM2  + M2*FF)
#define ACC_FLOATS (O_MCNT2 + M2)
// non-accumulated scratch:
#define O_X1IN    ACC_FLOATS
#define O_HA      (O_X1IN   + M1*DINP)
#define O_HB      (O_HA     + M1*HIDP)
#define O_X1PRE   (O_HB     + M1*HIDP)
#define O_M1MEAN  (O_X1PRE  + M1*FF)
#define O_X2IN    (O_M1MEAN + M1*FF)
#define O_X2PRE   (O_X2IN   + M2*DINP)
#define O_M2MEAN  (O_X2PRE  + M2*FF)
#define O_JSF1    (O_M2MEAN + M2*FF)
#define O_JSF2    (O_JSF1   + M1*JSFW)
#define WS_FLOATS (O_JSF2   + M2*JSFW)

// output layout (floats): r1, r2, x1, x2 concatenated
#define O_R1  0
#define O_R2  (O_R1 + NPIX*3)
#define O_X1  (O_R2 + NPIX*3)
#define O_X2  (O_X1 + M1*FF)

// ================= fused conv + segment accumulation =================
__global__ void conv_seg_kernel(const float* __restrict__ img,
                                const float* __restrict__ cw,
                                const float* __restrict__ cb,
                                const int*   __restrict__ cl1,
                                const int*   __restrict__ cl2,
                                float* x1sum, float* c1sum, float* mo1sum, float* cnt1,
                                float* c2sum, float* mo2sum, float* cntp2) {
    __shared__ float wl[27*FF];
    __shared__ float bl[FF];
    for (int t = threadIdx.x; t < 27*FF; t += blockDim.x) wl[t] = cw[t];
    for (int t = threadIdx.x; t < FF;    t += blockDim.x) bl[t] = cb[t];
    __syncthreads();

    int p = blockIdx.x * blockDim.x + threadIdx.x;
    if (p >= NPIX) return;

    int b   = p >> 16;
    int idx = p & 0xFFFF;
    int i   = idx >> 8;
    int j   = idx & 255;

    // gather 3x3x3 patch; SAME zero padding via clamp + mask (no branches)
    float v[27];
#pragma unroll
    for (int kh = 0; kh < 3; ++kh) {
#pragma unroll
        for (int kw = 0; kw < 3; ++kw) {
            int ih = i + kh - 1, jw = j + kw - 1;
            bool in = ((unsigned)ih < (unsigned)HH) && ((unsigned)jw < (unsigned)HH);
            int ihc = in ? ih : i;
            int jwc = in ? jw : j;
            float m = in ? 1.0f : 0.0f;
            long base = (((long)b*HH + ihc)*HH + jwc)*3;
#pragma unroll
            for (int ci = 0; ci < 3; ++ci)
                v[(kh*3+kw)*3+ci] = img[base+ci]*m;
        }
    }

    int c1 = cl1[p];
    float* xs = x1sum + (long)c1*FF;
    for (int f = 0; f < FF; ++f) {
        float a = bl[f];
#pragma unroll
        for (int k = 0; k < 27; ++k) a += v[k]*wl[k*FF+f];
        atomicAdd(xs+f, a);
    }

    float ci = (float)i * (1.0f/(float)HH);
    float cj = (float)j * (1.0f/(float)HH);
    atomicAdd(&c1sum[c1*2+0],  ci);
    atomicAdd(&c1sum[c1*2+1],  cj);
    atomicAdd(&mo1sum[c1*2+0], ci*ci);
    atomicAdd(&mo1sum[c1*2+1], cj*cj);
    atomicAdd(&cnt1[c1], 1.0f);

    int cr2 = cl2[c1];
    atomicAdd(&c2sum[cr2*2+0],  ci);
    atomicAdd(&c2sum[cr2*2+1],  cj);
    atomicAdd(&mo2sum[cr2*2+0], ci*ci);
    atomicAdd(&mo2sum[cr2*2+1], cj*cj);
    atomicAdd(&cntp2[cr2], 1.0f);
}

// ========== finalize level-1 means into (M1, DINP) zero-padded ==========
__global__ void finalize1_kernel(const float* x1sum, const float* c1sum,
                                 const float* mo1sum, const float* cnt1,
                                 float* x1in) {
    int r = blockIdx.x * blockDim.x + threadIdx.x;
    if (r >= M1) return;
    float inv = 1.0f / fmaxf(cnt1[r], 1.0f);
    float* o = x1in + (long)r*DINP;
    const float* s = x1sum + (long)r*FF;
    for (int c = 0; c < FF; ++c) o[c] = s[c]*inv;
    o[64] = c1sum[r*2+0]*inv;  o[65] = c1sum[r*2+1]*inv;
    o[66] = mo1sum[r*2+0]*inv; o[67] = mo1sum[r*2+1]*inv;
    for (int c = DIN; c < DINP; ++c) o[c] = 0.0f;
}

// ===== WMMA GEMM: C = [C +] A(M x Kpad) @ W(Kw x N) [+ bias] [relu] =====
// block = 128 threads (4 waves); block tile 64(M) x 32(N); K unconditional
// (A zero-padded to Kpad, W clamp+mask). Each wave: 16 rows x 2 col-subtiles.
__global__ void __launch_bounds__(128)
gemm_wmma_kernel(const float* __restrict__ A,
                 const float* __restrict__ W,
                 const float* __restrict__ bias,
                 float* __restrict__ C,
                 int M, int Kpad, int Kw, int N, int ldc, int Nstore,
                 int relu, int accf) {
    __shared__ __align__(16) _Float16 As[64*32];
    __shared__ __align__(16) _Float16 Bs[32*32];   // [colLocal][k]

    int tid  = threadIdx.x;
    int wave = tid >> 5;
    int lane = tid & 31;
    int half = lane >> 4;
    int l    = lane & 15;
    long rowBase = (long)blockIdx.x * 64;
    int  colBase = blockIdx.y * 32;

    v8f acc0 = {}, acc1 = {};

    for (int k0 = 0; k0 < Kpad; k0 += 32) {
        __syncthreads();
        // ---- stage A tile 64x32 (unconditional float4 loads) ----
#pragma unroll
        for (int i = 0; i < 4; ++i) {
            int g   = tid + i*128;        // 0..511 float4 groups
            int row = g >> 3;
            int c4  = (g & 7) << 2;
            const float* src = A + (rowBase + row)*Kpad + k0 + c4;
            float4 vv = *(const float4*)src;
            _Float16* d = &As[row*32 + c4];
            d[0] = (_Float16)vv.x; d[1] = (_Float16)vv.y;
            d[2] = (_Float16)vv.z; d[3] = (_Float16)vv.w;
        }
        // ---- stage B tile 32(K)x32(N), transposed, clamp+mask ----
#pragma unroll
        for (int i = 0; i < 8; ++i) {
            int g   = tid + i*128;        // 0..1023
            int k   = g >> 5;
            int cl  = g & 31;
            int col = colBase + cl;
            int kk  = k0 + k;
            bool ok = (kk < Kw) && (col < N);
            int kc  = ok ? kk  : (Kw - 1);
            int cc  = ok ? col : 0;
            float bv = W[(long)kc*N + cc];
            Bs[cl*32 + k] = (_Float16)(ok ? bv : 0.0f);
        }
        __syncthreads();

        // ---- per-lane fragments from LDS (16B aligned vector reads) ----
        const _Float16* ap = &As[(wave*16 + l)*32];
        v8h a_lo = *(const v8h*)(ap + half*8);
        v8h a_hi = *(const v8h*)(ap + 16 + half*8);
        v16h a = __builtin_shufflevector(a_lo, a_hi,
                 0,1,2,3,4,5,6,7,8,9,10,11,12,13,14,15);

        const _Float16* bp0 = &Bs[l*32 + half*8];
        v8h b0l = *(const v8h*)bp0;
        v8h b0h = *(const v8h*)(bp0 + 16);
        v16h b0 = __builtin_shufflevector(b0l, b0h,
                  0,1,2,3,4,5,6,7,8,9,10,11,12,13,14,15);

        const _Float16* bp1 = &Bs[(16 + l)*32 + half*8];
        v8h b1l = *(const v8h*)bp1;
        v8h b1h = *(const v8h*)(bp1 + 16);
        v16h b1 = __builtin_shufflevector(b1l, b1h,
                  0,1,2,3,4,5,6,7,8,9,10,11,12,13,14,15);

        acc0 = __builtin_amdgcn_wmma_f32_16x16x32_f16(
                   false, a, false, b0, (short)0, acc0, false, false);
        acc1 = __builtin_amdgcn_wmma_f32_16x16x32_f16(
                   false, a, false, b1, (short)0, acc1, false, false);
    }

    // ---- epilogue ----
    int col0 = colBase + l;
    int col1 = colBase + 16 + l;
    float bias0 = 0.0f, bias1 = 0.0f;
    if (bias) {
        bias0 = bias[col0 < N ? col0 : (N-1)];
        bias1 = bias[col1 < N ? col1 : (N-1)];
    }
    if (col0 >= N) bias0 = 0.0f;
    if (col1 >= N) bias1 = 0.0f;

#pragma unroll
    for (int v = 0; v < 8; ++v) {
        long row = rowBase + wave*16 + v + 8*half;
        if (col0 < Nstore) {
            float x = acc0[v];
            if (accf) x += C[row*ldc + col0];
            x += bias0;
            if (relu) x = fmaxf(x, 0.0f);
            C[row*ldc + col0] = (col0 < N) ? x : 0.0f;
        }
        if (col1 < Nstore) {
            float x = acc1[v];
            if (accf) x += C[row*ldc + col1];
            x += bias1;
            if (relu) x = fmaxf(x, 0.0f);
            C[row*ldc + col1] = (col1 < N) ? x : 0.0f;
        }
    }
}

// ================= graph-conv neighbor accumulation =====================
__global__ void edge_acc_kernel(const float* __restrict__ x,
                                const int* __restrict__ edges, int E,
                                float* msum, float* mcnt) {
    int t = blockIdx.x * blockDim.x + threadIdx.x;
    if (t >= E*8) return;
    int e = t >> 3, ch = t & 7;
    int src = edges[e];
    int dst = edges[E + e];
    const float* xr = x    + (long)src*FF + ch*8;
    float*       ms = msum + (long)dst*FF + ch*8;
#pragma unroll
    for (int c = 0; c < 8; ++c) atomicAdd(ms+c, xr[c]);
    if (ch == 0) atomicAdd(mcnt+dst, 1.0f);
}

__global__ void seg_div_kernel(const float* s, const float* cnt, float* o, int R) {
    int t = blockIdx.x * blockDim.x + threadIdx.x;
    if (t >= R*FF) return;
    int r = t >> 6;
    o[t] = s[t] / fmaxf(cnt[r], 1.0f);
}

// ================= x2 row pooling (cluster2 over M1 rows) ===============
__global__ void x2row_acc_kernel(const float* __restrict__ x1,
                                 const int* __restrict__ cl2,
                                 float* x2sum, float* cntr2) {
    int t = blockIdx.x * blockDim.x + threadIdx.x;
    if (t >= M1*8) return;
    int r = t >> 3, ch = t & 7;
    int seg = cl2[r];
    const float* xr = x1    + (long)r*FF   + ch*8;
    float*       xs = x2sum + (long)seg*FF + ch*8;
#pragma unroll
    for (int c = 0; c < 8; ++c) atomicAdd(xs+c, xr[c]);
    if (ch == 0) atomicAdd(cntr2+seg, 1.0f);
}

__global__ void finalize2_kernel(const float* x2sum, const float* cntr2,
                                 const float* c2sum, const float* mo2sum,
                                 const float* cntp2, float* x2in) {
    int r = blockIdx.x * blockDim.x + threadIdx.x;
    if (r >= M2) return;
    float invr = 1.0f / fmaxf(cntr2[r], 1.0f);
    float* o = x2in + (long)r*DINP;
    const float* s = x2sum + (long)r*FF;
    for (int c = 0; c < FF; ++c) o[c] = s[c]*invr;
    float invp = 1.0f / fmaxf(cntp2[r], 1.0f);
    o[64] = c2sum[r*2+0]*invp;  o[65] = c2sum[r*2+1]*invp;
    o[66] = mo2sum[r*2+0]*invp; o[67] = mo2sum[r*2+1]*invp;
    for (int c = DIN; c < DINP; ++c) o[c] = 0.0f;
}

// copy centroid columns into jsf cols 18,19 (reference concat order)
__global__ void cen_copy_kernel(const float* xin, float* jsf, int R) {
    int r = blockIdx.x * blockDim.x + threadIdx.x;
    if (r >= R) return;
    jsf[(long)r*JSFW + 18] = xin[(long)r*DINP + 64];
    jsf[(long)r*JSFW + 19] = xin[(long)r*DINP + 65];
}

// ================= quadratic texture render =============================
__global__ void render_kernel(const float* __restrict__ jsf,
                              const int* __restrict__ cl1,
                              const int* __restrict__ cl2,   // null -> level 1
                              float* __restrict__ out) {
    int p = blockIdx.x * blockDim.x + threadIdx.x;
    if (p >= NPIX) return;
    int idx = p & 0xFFFF;
    int i = idx >> 8, j = idx & 255;
    float gx = (float)i * (1.0f/(float)HH);
    float gy = (float)j * (1.0f/(float)HH);
    int c = cl1[p];
    if (cl2) c = cl2[c];
    const float* f = jsf + (long)c*JSFW;
    float dx = gx - f[0];
    float dy = gy - f[1];
#pragma unroll
    for (int k = 0; k < 3; ++k) {
        const float* pa = f + 2 + k*6;
        out[(long)p*3 + k] = pa[0] + pa[1]*dx + pa[2]*dy
                           + pa[3]*dx*dx + pa[4]*dy*dy + pa[5]*dx*dy;
    }
}

// ================= host orchestration ===================================
static inline void launch_gemm(const float* A, const float* W, const float* bias,
                               float* C, int M, int Kpad, int Kw, int N,
                               int ldc, int Nstore, int relu, int accf,
                               hipStream_t s) {
    dim3 grid(M/64, (Nstore + 31)/32);
    gemm_wmma_kernel<<<grid, 128, 0, s>>>(A, W, bias, C, M, Kpad, Kw, N,
                                          ldc, Nstore, relu, accf);
}

extern "C" void kernel_launch(void* const* d_in, const int* in_sizes, int n_in,
                              void* d_out, int out_size, void* d_ws, size_t ws_size,
                              hipStream_t stream) {
    const float* img    = (const float*)d_in[0];
    const float* conv_w = (const float*)d_in[1];
    const float* conv_b = (const float*)d_in[2];
    const float* t1_wi  = (const float*)d_in[3];
    const float* t1_bi  = (const float*)d_in[4];
    const float* t1_wh  = (const float*)d_in[5];
    const float* t1_bh  = (const float*)d_in[6];
    const float* t1_wo  = (const float*)d_in[7];
    const float* t1_bo  = (const float*)d_in[8];
    const float* t2_wi  = (const float*)d_in[9];
    const float* t2_bi  = (const float*)d_in[10];
    const float* t2_wh  = (const float*)d_in[11];
    const float* t2_bh  = (const float*)d_in[12];
    const float* t2_wo  = (const float*)d_in[13];
    const float* t2_bo  = (const float*)d_in[14];
    const float* q_wi   = (const float*)d_in[15];
    const float* q_bi   = (const float*)d_in[16];
    const float* q_wh   = (const float*)d_in[17];
    const float* q_bh   = (const float*)d_in[18];
    const float* q_wo   = (const float*)d_in[19];
    const float* q_bo   = (const float*)d_in[20];
    const float* gc1_wn = (const float*)d_in[21];
    const float* gc1_ws = (const float*)d_in[22];
    const float* gc1_b  = (const float*)d_in[23];
    const float* gc2_wn = (const float*)d_in[24];
    const float* gc2_ws = (const float*)d_in[25];
    const float* gc2_b  = (const float*)d_in[26];
    const int* cluster1 = (const int*)d_in[27];
    const int* cluster2 = (const int*)d_in[28];
    const int* edges1   = (const int*)d_in[29];
    const int* edges2   = (const int*)d_in[30];

    float* ws  = (float*)d_ws;
    float* out = (float*)d_out;

    float* x1sum  = ws + O_X1SUM;
    float* c1sum  = ws + O_C1SUM;
    float* mo1sum = ws + O_MO1SUM;
    float* cnt1   = ws + O_CNT1;
    float* c2sum  = ws + O_C2SUM;
    float* mo2sum = ws + O_MO2SUM;
    float* cntp2  = ws + O_CNTP2;
    float* msum1  = ws + O_MSUM1;
    float* mcnt1  = ws + O_MCNT1;
    float* x2sum  = ws + O_X2SUM;
    float* cntr2  = ws + O_CNTR2;
    float* msum2  = ws + O_MSUM2;
    float* mcnt2  = ws + O_MCNT2;
    float* x1in   = ws + O_X1IN;
    float* ha     = ws + O_HA;
    float* hb     = ws + O_HB;
    float* x1pre  = ws + O_X1PRE;
    float* m1mean = ws + O_M1MEAN;
    float* x2in   = ws + O_X2IN;
    float* x2pre  = ws + O_X2PRE;
    float* m2mean = ws + O_M2MEAN;
    float* jsf1   = ws + O_JSF1;
    float* jsf2   = ws + O_JSF2;

    float* r1  = out + O_R1;
    float* r2  = out + O_R2;
    float* x1o = out + O_X1;
    float* x2o = out + O_X2;

    // zero all atomic accumulators (graph-capture-safe async memset)
    hipMemsetAsync(d_ws, 0, (size_t)ACC_FLOATS * sizeof(float), stream);

    // 1) fused conv 3x3 (3->64) + all pixel-level segment sums
    conv_seg_kernel<<<NPIX/256, 256, 0, stream>>>(img, conv_w, conv_b,
        cluster1, cluster2, x1sum, c1sum, mo1sum, cnt1, c2sum, mo2sum, cntp2);

    // 2) level-1 means -> x1in (M1 x DINP, zero padded)
    finalize1_kernel<<<M1/256, 256, 0, stream>>>(x1sum, c1sum, mo1sum, cnt1, x1in);

    // 3) t1 FC block: 68 -> 100 -> 100 -> 100 -> 64
    launch_gemm(x1in, t1_wi, t1_bi, ha, M1, DINP, DIN, HID, HIDP, HIDP, 1, 0, stream);
    launch_gemm(ha, t1_wh,           t1_bh,     hb, M1, HIDP, HID, HID, HIDP, HIDP, 1, 0, stream);
    launch_gemm(hb, t1_wh + HID*HID, t1_bh+HID, ha, M1, HIDP, HID, HID, HIDP, HIDP, 1, 0, stream);
    launch_gemm(ha, t1_wo, t1_bo, x1pre, M1, HIDP, HID, FF, FF, FF, 0, 0, stream);

    // 4) graph conv 1: m = segmean(x1pre[e0] over e1); x1 = m@Wn + x@Ws + b
    edge_acc_kernel<<<(E1*8)/256, 256, 0, stream>>>(x1pre, edges1, E1, msum1, mcnt1);
    seg_div_kernel<<<(M1*FF)/256, 256, 0, stream>>>(msum1, mcnt1, m1mean, M1);
    launch_gemm(m1mean, gc1_wn, gc1_b, x1o, M1, FF, FF, FF, FF, FF, 0, 0, stream);
    launch_gemm(x1pre,  gc1_ws, nullptr, x1o, M1, FF, FF, FF, FF, FF, 0, 1, stream);

    // 5) level-2 pooling: rows of x1 by cluster2; coords pooled in step 1
    x2row_acc_kernel<<<(M1*8)/256, 256, 0, stream>>>(x1o, cluster2, x2sum, cntr2);
    finalize2_kernel<<<(M2+255)/256, 256, 0, stream>>>(x2sum, cntr2, c2sum, mo2sum, cntp2, x2in);

    // 6) t2 FC block
    launch_gemm(x2in, t2_wi, t2_bi, ha, M2, DINP, DIN, HID, HIDP, HIDP, 1, 0, stream);
    launch_gemm(ha, t2_wh,           t2_bh,     hb, M2, HIDP, HID, HID, HIDP, HIDP, 1, 0, stream);
    launch_gemm(hb, t2_wh + HID*HID, t2_bh+HID, ha, M2, HIDP, HID, HID, HIDP, HIDP, 1, 0, stream);
    launch_gemm(ha, t2_wo, t2_bo, x2pre, M2, HIDP, HID, FF, FF, FF, 0, 0, stream);

    // 7) graph conv 2
    edge_acc_kernel<<<(E2*8)/256, 256, 0, stream>>>(x2pre, edges2, E2, msum2, mcnt2);
    seg_div_kernel<<<(M2*FF)/256, 256, 0, stream>>>(msum2, mcnt2, m2mean, M2);
    launch_gemm(m2mean, gc2_wn, gc2_b, x2o, M2, FF, FF, FF, FF, FF, 0, 0, stream);
    launch_gemm(x2pre,  gc2_ws, nullptr, x2o, M2, FF, FF, FF, FF, FF, 0, 1, stream);

    // 8) q FC block on x1 -> jsf1[:, :18]; append cen1
    launch_gemm(x1o, q_wi, q_bi, ha, M1, FF, FF, HID, HIDP, HIDP, 1, 0, stream);
    launch_gemm(ha, q_wh,           q_bh,     hb, M1, HIDP, HID, HID, HIDP, HIDP, 1, 0, stream);
    launch_gemm(hb, q_wh + HID*HID, q_bh+HID, ha, M1, HIDP, HID, HID, HIDP, HIDP, 1, 0, stream);
    launch_gemm(ha, q_wo, q_bo, jsf1, M1, HIDP, HID, 18, JSFW, 18, 0, 0, stream);
    cen_copy_kernel<<<(M1+255)/256, 256, 0, stream>>>(x1in, jsf1, M1);

    // 9) q FC block on x2 -> jsf2[:, :18]; append cen2
    launch_gemm(x2o, q_wi, q_bi, ha, M2, FF, FF, HID, HIDP, HIDP, 1, 0, stream);
    launch_gemm(ha, q_wh,           q_bh,     hb, M2, HIDP, HID, HID, HIDP, HIDP, 1, 0, stream);
    launch_gemm(hb, q_wh + HID*HID, q_bh+HID, ha, M2, HIDP, HID, HID, HIDP, HIDP, 1, 0, stream);
    launch_gemm(ha, q_wo, q_bo, jsf2, M2, HIDP, HID, 18, JSFW, 18, 0, 0, stream);
    cen_copy_kernel<<<(M2+255)/256, 256, 0, stream>>>(x2in, jsf2, M2);

    // 10) render both levels
    render_kernel<<<NPIX/256, 256, 0, stream>>>(jsf1, cluster1, nullptr,  r1);
    render_kernel<<<NPIX/256, 256, 0, stream>>>(jsf2, cluster1, cluster2, r2);
}
